// ResidualSAGEBlock_2697239461992
// MI455X (gfx1250) — matrix operand order repaired and checked by reading, not compiled, not run on hardware
//
#include <hip/hip_runtime.h>
#include <hip/hip_bf16.h>
#include <math.h>

#define NNODES 100000
#define NEDGES 640000
#define INF    128
#define OUTF   256

typedef __attribute__((ext_vector_type(16))) __bf16 v16bf;
typedef __attribute__((ext_vector_type(8)))  float  v8f;

union V16 { v16bf v; unsigned int u[8]; };

__device__ __forceinline__ unsigned short f2bf(float f) {
    unsigned int u = __float_as_uint(f);
    unsigned int r = u + 0x7FFFu + ((u >> 16) & 1u);   // round-to-nearest-even
    return (unsigned short)(r >> 16);
}
__device__ __forceinline__ unsigned int pack2(float a, float b) {
    return (unsigned int)f2bf(a) | ((unsigned int)f2bf(b) << 16);
}

// ---------------- weight f32 -> bf16 conversion (runs once per launch) ----------
__global__ void cvt_weights(const float* __restrict__ wl, const float* __restrict__ wr,
                            const float* __restrict__ wp,
                            unsigned short* __restrict__ ol, unsigned short* __restrict__ orr,
                            unsigned short* __restrict__ op) {
    int i = blockIdx.x * 256 + threadIdx.x;
    if (i < OUTF * INF) {
        ol[i]  = f2bf(wl[i]);
        orr[i] = f2bf(wr[i]);
        op[i]  = f2bf(wp[i]);
    }
}

// ---------------- edge scatter: summed[dst] += x[src]; cnt[dst] += 1 ------------
__global__ __launch_bounds__(256) void edge_scatter(const float* __restrict__ x,
                                                    const int* __restrict__ src,
                                                    const int* __restrict__ dst,
                                                    float* __restrict__ summed,
                                                    float* __restrict__ cnt) {
    int idx = blockIdx.x * 256 + threadIdx.x;          // NEDGES*32 threads
    if (idx >= NEDGES * 32) return;
    int e = idx >> 5;
    int c = (idx & 31) * 4;
    int s = src[e], d = dst[e];
    float4 v = *(const float4*)(x + (size_t)s * INF + c);
    float* p = summed + (size_t)d * INF + c;
    atomicAdd(p + 0, v.x);
    atomicAdd(p + 1, v.y);
    atomicAdd(p + 2, v.z);
    atomicAdd(p + 3, v.w);
    if ((idx & 31) == 0) atomicAdd(cnt + d, 1.0f);
}

// ---------------- fragment loaders (wave32 WMMA bf16 layouts, ISA 7.12.2) -------
// A 16x32 bf16: lanes 0-15 -> M=lane, K = kk*32 + {0..7, 16..23}
//               lanes 16-31 -> M=lane-16, K = kk*32 + {8..15, 24..31}
__device__ __forceinline__ v16bf load_a_frag(const unsigned int (*tile)[INF / 2],
                                             int lane, int kk) {
    V16 f;
    int row  = lane & 15;
    int off  = (lane >> 4) ? 4 : 0;       // +8 bf16 = +4 uints
    int base = kk * 16 + off;             // kk*32 bf16 cols = kk*16 uints
#pragma unroll
    for (int t = 0; t < 4; ++t) {
        f.u[t]     = tile[row][base + t];
        f.u[4 + t] = tile[row][base + 8 + t];  // +16 bf16 cols = +8 uints
    }
    return f.v;
}

// B 32x16 bf16 (B = W^T): lane%16 = output column n, lanes 0-15 K=kk*32+{0..7,16..23},
// lanes 16-31 K=kk*32+{8..15,24..31}.  W is row-major [OUTF][INF] so B[k][n] = W[n][k].
__device__ __forceinline__ v16bf load_b_frag(const unsigned short* __restrict__ W,
                                             int lane, int j, int kk) {
    V16 f;
    int n  = lane & 15;
    int k0 = kk * 32 + ((lane >> 4) ? 8 : 0);
    const uint4* p = (const uint4*)(W + (size_t)(j * 16 + n) * INF + k0);
    uint4 a = p[0], b = p[2];
    f.u[0] = a.x; f.u[1] = a.y; f.u[2] = a.z; f.u[3] = a.w;
    f.u[4] = b.x; f.u[5] = b.y; f.u[6] = b.z; f.u[7] = b.w;
    return f.v;
}

// ---------------- fused node kernel: 2 waves/block, 1 wave = 16-node tile -------
__global__ __launch_bounds__(64) void node_fused(const float* __restrict__ x,
                                                 const float* __restrict__ summed,
                                                 const float* __restrict__ cnt,
                                                 const unsigned short* __restrict__ wl,
                                                 const unsigned short* __restrict__ wr,
                                                 const unsigned short* __restrict__ wp,
                                                 const float* __restrict__ bl,
                                                 const float* __restrict__ gamma,
                                                 const float* __restrict__ beta,
                                                 float* __restrict__ out) {
    __shared__ unsigned int xa[2][16][INF / 2];   // x tile, packed bf16
    __shared__ unsigned int ag[2][16][INF / 2];   // agg tile, packed bf16
    __shared__ float        hb[2][16][OUTF];      // f32 h staging for LayerNorm
    __shared__ float        mu_s[2][16];
    __shared__ float        rs_s[2][16];

    const int wave  = threadIdx.x >> 5;
    const int lane  = threadIdx.x & 31;
    const int tile  = blockIdx.x * 2 + wave;
    const int node0 = tile * 16;

    // ---- stage x / agg tiles into LDS as bf16 ----
    for (int r = 0; r < 16; ++r) {
        int node = node0 + r;
        int c    = lane * 4;
        float4 xv = make_float4(0.f, 0.f, 0.f, 0.f);
        float4 sv = make_float4(0.f, 0.f, 0.f, 0.f);
        float  rcp = 0.f;
        if (node < NNODES) {
            xv  = *(const float4*)(x + (size_t)node * INF + c);
            sv  = *(const float4*)(summed + (size_t)node * INF + c);
            rcp = 1.0f / fmaxf(cnt[node], 1.0f);
        }
        xa[wave][r][lane * 2 + 0] = pack2(xv.x, xv.y);
        xa[wave][r][lane * 2 + 1] = pack2(xv.z, xv.w);
        ag[wave][r][lane * 2 + 0] = pack2(sv.x * rcp, sv.y * rcp);
        ag[wave][r][lane * 2 + 1] = pack2(sv.z * rcp, sv.w * rcp);
    }

    const int n  = lane & 15;
    const int mo = (lane >> 4) ? 8 : 0;

    // ---- pass 1: h = agg @ W_l^T + x @ W_r^T + b_l  -> LDS ----
    for (int j = 0; j < 16; ++j) {
        v8f acc = {};
#pragma unroll
        for (int kk = 0; kk < 4; ++kk) {
            v16bf aa = load_a_frag(ag[wave], lane, kk);
            v16bf ba = load_b_frag(wl, lane, j, kk);
            acc = __builtin_amdgcn_wmma_f32_16x16x32_bf16(false, aa, false, ba,
                                                          (short)0, acc, false, false);
            v16bf ax = load_a_frag(xa[wave], lane, kk);
            v16bf bx = load_b_frag(wr, lane, j, kk);
            acc = __builtin_amdgcn_wmma_f32_16x16x32_bf16(false, ax, false, bx,
                                                          (short)0, acc, false, false);
        }
        float bias = bl[j * 16 + n];
#pragma unroll
        for (int i = 0; i < 8; ++i)
            hb[wave][i + mo][j * 16 + n] = acc[i] + bias;
    }

    // ---- LayerNorm stats over 256 features per row ----
    {
        int row  = lane & 15;
        int half = lane >> 4;
        float s = 0.f, ss = 0.f;
#pragma unroll 4
        for (int c = 0; c < 128; ++c) {
            float v = hb[wave][row][half * 128 + c];
            s  += v;
            ss += v * v;
        }
        s  += __shfl_xor(s, 16, 32);
        ss += __shfl_xor(ss, 16, 32);
        float mu  = s * (1.0f / 256.0f);
        float var = ss * (1.0f / 256.0f) - mu * mu;
        float rst = rsqrtf(var + 1e-5f);
        if (lane < 16) { mu_s[wave][row] = mu; rs_s[wave][row] = rst; }
    }

    // ---- pass 2: r = x @ W_proj^T ; out = GELU(LN(h)) + r ----
    for (int j = 0; j < 16; ++j) {
        v8f acc = {};
#pragma unroll
        for (int kk = 0; kk < 4; ++kk) {
            v16bf ax = load_a_frag(xa[wave], lane, kk);
            v16bf bp = load_b_frag(wp, lane, j, kk);
            acc = __builtin_amdgcn_wmma_f32_16x16x32_bf16(false, ax, false, bp,
                                                          (short)0, acc, false, false);
        }
        int col = j * 16 + n;
        float g  = gamma[col];
        float be = beta[col];
#pragma unroll
        for (int i = 0; i < 8; ++i) {
            int m    = i + mo;
            int node = node0 + m;
            if (node < NNODES) {
                float h = hb[wave][m][col];
                float v = (h - mu_s[wave][m]) * rs_s[wave][m] * g + be;
                float ge = 0.5f * v * (1.0f + erff(v * 0.70710678118654752f));
                out[(size_t)node * OUTF + col] = ge + acc[i];
            }
        }
    }
}

extern "C" void kernel_launch(void* const* d_in, const int* in_sizes, int n_in,
                              void* d_out, int out_size, void* d_ws, size_t ws_size,
                              hipStream_t stream) {
    const float* x      = (const float*)d_in[0];
    const int*   ei     = (const int*)d_in[1];
    const float* W_l    = (const float*)d_in[2];
    const float* b_l    = (const float*)d_in[3];
    const float* W_r    = (const float*)d_in[4];
    const float* gamma  = (const float*)d_in[5];
    const float* beta   = (const float*)d_in[6];
    const float* W_proj = (const float*)d_in[7];
    float*       out    = (float*)d_out;

    const int* src = ei;
    const int* dst = ei + NEDGES;

    // workspace carve-up
    float* summed = (float*)d_ws;                        // N*128 f32
    float* cnt    = summed + (size_t)NNODES * INF;       // N f32
    unsigned short* wlb = (unsigned short*)(cnt + NNODES);
    unsigned short* wrb = wlb + OUTF * INF;
    unsigned short* wpb = wrb + OUTF * INF;

    hipMemsetAsync(summed, 0, ((size_t)NNODES * INF + NNODES) * sizeof(float), stream);

    cvt_weights<<<(OUTF * INF + 255) / 256, 256, 0, stream>>>(W_l, W_r, W_proj,
                                                              wlb, wrb, wpb);

    edge_scatter<<<(NEDGES * 32 + 255) / 256, 256, 0, stream>>>(x, src, dst,
                                                                summed, cnt);

    const int ntiles  = (NNODES + 15) / 16;              // 6250
    const int nblocks = (ntiles + 1) / 2;                // 3125 (2 waves/block)
    node_fused<<<nblocks, 64, 0, stream>>>(x, summed, cnt, wlb, wrb, wpb,
                                           b_l, gamma, beta, out);
}
